// DisentangledSelfAttention_56092272886441
// MI455X (gfx1250) — matrix operand order repaired and checked by reading, not compiled, not run on hardware
//
#include <hip/hip_runtime.h>

// DeBERTa disentangled self-attention for MI455X (gfx1250), wave32 + WMMA bf16.
// Requires ws_size >= ~71 MB.

typedef __bf16 bf16;
typedef bf16  v16bf __attribute__((ext_vector_type(16)));
typedef float v8f   __attribute__((ext_vector_type(8)));

#define HID_ 1024
#define DH_  64
#define NH_  16
#define NB_  4
#define SEQ_ 1024
#define TK_  1024   // 2*K relative positions
#define EPS_ 1e-7f

__device__ __constant__ const float SCALE_F = 0.07216878364870323f; // 1/sqrt(3*64)

// ---------------------------------------------------------------- WMMA helpers

static __device__ __forceinline__ v8f wmma_bf16(v16bf a, v16bf b, v8f c) {
  return __builtin_amdgcn_wmma_f32_16x16x32_bf16(false, a, false, b, (short)0, c,
                                                 false, false);
}

static __device__ __forceinline__ v8f zero8() {
  v8f z = {0.f, 0.f, 0.f, 0.f, 0.f, 0.f, 0.f, 0.f};
  return z;
}

// A-matrix 16x32 bf16 fragment, row-major source: A[m][k] = base[m*ld + k].
// Lane l holds row m=l&15; element e<8 -> k=(l>>4)*8+e ; e>=8 -> k=16+(l>>4)*8+(e-8).
static __device__ __forceinline__ v16bf load_a_frag(const bf16* __restrict__ base, int ld) {
  int l = threadIdx.x & 31;
  const bf16* p = base + (size_t)(l & 15) * ld + ((l >> 4) * 8);
  v16bf a;
#pragma unroll
  for (int e = 0; e < 8; ++e) a[e] = p[e];
#pragma unroll
  for (int e = 0; e < 8; ++e) a[8 + e] = p[16 + e];
  return a;
}

// A fragment with clamped row index (for rel_q diagonal band).
static __device__ __forceinline__ v16bf load_a_frag_clamp(const bf16* __restrict__ mat, int ld,
                                                          int rowbase, int rowmax, int koff) {
  int l = threadIdx.x & 31;
  int row = rowbase + (l & 15);
  row = row < 0 ? 0 : (row > rowmax ? rowmax : row);
  const bf16* p = mat + (size_t)row * ld + koff + ((l >> 4) * 8);
  v16bf a;
#pragma unroll
  for (int e = 0; e < 8; ++e) a[e] = p[e];
#pragma unroll
  for (int e = 0; e < 8; ++e) a[8 + e] = p[16 + e];
  return a;
}

// B-matrix 32x16 bf16 fragment from a TRANSPOSED source: B[k][n] = base[n*ld + k].
// Lane l holds column n=l&15; element e<8 -> k=(l>>4)*8+e ; e>=8 -> k=16+(l>>4)*8+(e-8).
static __device__ __forceinline__ v16bf load_bt_frag(const bf16* __restrict__ base, int ld) {
  int l = threadIdx.x & 31;
  const bf16* p = base + (size_t)(l & 15) * ld + ((l >> 4) * 8);
  v16bf b;
#pragma unroll
  for (int e = 0; e < 8; ++e) b[e] = p[e];
#pragma unroll
  for (int e = 0; e < 8; ++e) b[8 + e] = p[16 + e];
  return b;
}

// Same, with clamped row index (for rel_k diagonal band).
static __device__ __forceinline__ v16bf load_bt_frag_clamp(const bf16* __restrict__ mat, int ld,
                                                           int rowbase, int rowmax, int koff) {
  int l = threadIdx.x & 31;
  int row = rowbase + (l & 15);
  row = row < 0 ? 0 : (row > rowmax ? rowmax : row);
  const bf16* p = mat + (size_t)row * ld + koff + ((l >> 4) * 8);
  v16bf b;
#pragma unroll
  for (int e = 0; e < 8; ++e) b[e] = p[e];
#pragma unroll
  for (int e = 0; e < 8; ++e) b[8 + e] = p[16 + e];
  return b;
}

// B-matrix 32x16 fragment from ROW-MAJOR source: B[k][n] = base[k*ld + n]  (V tiles).
static __device__ __forceinline__ v16bf load_b_rowmajor_frag(const bf16* __restrict__ base, int ld) {
  int l = threadIdx.x & 31;
  const bf16* p = base + (l & 15) + (size_t)((l >> 4) * 8) * ld;
  v16bf b;
#pragma unroll
  for (int e = 0; e < 8; ++e) b[e] = p[(size_t)e * ld];
#pragma unroll
  for (int e = 0; e < 8; ++e) b[8 + e] = p[(size_t)(16 + e) * ld];
  return b;
}

// ------------------------------------------------------------------ prep kernels

__global__ __launch_bounds__(256) void dsa_cvt_bf16(const float* __restrict__ in,
                                                    bf16* __restrict__ out, int n) {
  int i = blockIdx.x * 256 + threadIdx.x;
  if (i < n) out[i] = (bf16)in[i];
}

// W [HID,HID] f32 row-major -> WT [out=HID, in=HID] bf16:  WT[n][k] = W[k][n]
__global__ __launch_bounds__(256) void dsa_transpose_bf16(const float* __restrict__ in,
                                                          bf16* __restrict__ out) {
  int n = blockIdx.x;
  for (int k = threadIdx.x; k < HID_; k += 256)
    out[(size_t)n * HID_ + k] = (bf16)in[(size_t)k * HID_ + n];
}

// ------------------------------------------------------------------ WMMA GEMM
// out[M, HID] = A[M, HID](bf16) @ WT^T + bias (+ residual). 4 waves/block,
// wave w computes C tile (blockIdx.x*16, (blockIdx.y*4+w)*16).

__global__ __launch_bounds__(128)
void dsa_gemm_wmma(const bf16* __restrict__ A, const bf16* __restrict__ WT,
                   const float* __restrict__ bias, const float* __restrict__ residual,
                   float* __restrict__ outF, bf16* __restrict__ outB) {
  int wave = threadIdx.x >> 5;
  int l    = threadIdx.x & 31;
  int m0   = blockIdx.x * 16;
  int n0   = (blockIdx.y * 4 + wave) * 16;

  v8f acc = zero8();
#pragma unroll 4
  for (int k0 = 0; k0 < HID_; k0 += 32) {
    v16bf a = load_a_frag(A + (size_t)m0 * HID_ + k0, HID_);
    v16bf b = load_bt_frag(WT + (size_t)n0 * HID_ + k0, HID_);
    acc = wmma_bf16(a, b, acc);
  }

  int n  = n0 + (l & 15);
  int mh = (l >> 4) * 8;
  float bn = bias[n];
#pragma unroll
  for (int r = 0; r < 8; ++r) {
    int row = m0 + mh + r;
    float v = acc[r] + bn;
    if (residual) v += residual[(size_t)row * HID_ + n];
    if (outF) outF[(size_t)row * HID_ + n] = v;
    else      outB[(size_t)row * HID_ + n] = (bf16)v;
  }
}

// --------------------------------------------------------------- fused attention
// One block per (b, h, 64-row chunk); wave owns a 16-row i-tile, streams j by 32.
// S[i,j] = SCALE*( q_i.k_j + q_i.rel_k[c] + rel_q[c].k_j ),  c = clip(i-j+512, 0, 1023).
// Within an (i-tile, j-subtile) pair c spans a 31-wide band -> band GEMMs on WMMA,
// then per-element gather via LDS. Online softmax; P@V accumulated with WMMA.

__global__ __launch_bounds__(128)
void dsa_attn_wmma(const bf16* __restrict__ Qh, const bf16* __restrict__ Kh,
                   const bf16* __restrict__ Vh, const bf16* __restrict__ Qr,
                   const bf16* __restrict__ Kr, bf16* __restrict__ Oattn) {
  __shared__ float lsT[4][16][66];   // cqrk band tiles, 2 subtiles x 32 cols
  __shared__ float lsU[4][64][17];   // rqck band tiles, 2 subtiles x 32 rows
  __shared__ bf16  lsP[4][16][32];   // probabilities (C-layout -> A-layout bridge)

  int wave = threadIdx.x >> 5;
  int l    = threadIdx.x & 31;
  int bid  = blockIdx.x;
  int b    = bid / (NH_ * (SEQ_ / 64));
  int rem  = bid % (NH_ * (SEQ_ / 64));
  int h    = rem / (SEQ_ / 64);
  int ic   = rem % (SEQ_ / 64);
  int i0   = ic * 64 + wave * 16;

  const bf16* Qb  = Qh + (size_t)b * SEQ_ * HID_ + h * DH_;
  const bf16* Kb  = Kh + (size_t)b * SEQ_ * HID_ + h * DH_;
  const bf16* Vb  = Vh + (size_t)b * SEQ_ * HID_ + h * DH_;
  const bf16* Qrh = Qr + h * DH_;
  const bf16* Krh = Kr + h * DH_;

  // Q row fragments (reused for every j step): k-steps 0 and 32 of DH=64.
  v16bf aQ0 = load_a_frag(Qb + (size_t)i0 * HID_, HID_);
  v16bf aQ1 = load_a_frag(Qb + (size_t)i0 * HID_ + 32, HID_);

  v8f oacc[4];
#pragma unroll
  for (int t = 0; t < 4; ++t) oacc[t] = zero8();
  float rmax[8], rsum[8];
#pragma unroll
  for (int r = 0; r < 8; ++r) { rmax[r] = -1e30f; rsum[r] = 0.f; }

  int nn = l & 15;
  int mh = (l >> 4) * 8;

  for (int j0 = 0; j0 < SEQ_; j0 += 32) {
    v8f S[2];
    S[0] = zero8(); S[1] = zero8();

#pragma unroll
    for (int s = 0; s < 2; ++s) {
      int js = j0 + 16 * s;
      // K^T fragments for this j-subtile (shared by main score and rqck term)
      v16bf bK0 = load_bt_frag(Kb + (size_t)js * HID_, HID_);
      v16bf bK1 = load_bt_frag(Kb + (size_t)js * HID_ + 32, HID_);
      S[s] = wmma_bf16(aQ0, bK0, S[s]);
      S[s] = wmma_bf16(aQ1, bK1, S[s]);

      int bandbase = i0 - js + 497;   // c' = (i-i0) - (j-js) + 15, raw = bandbase + c'

      // content->position: T[m][c'] = q_i . rel_k[clamp(bandbase+c')]
      v8f T0 = zero8(), T1 = zero8();
      T0 = wmma_bf16(aQ0, load_bt_frag_clamp(Krh, HID_, bandbase,      TK_ - 1, 0),  T0);
      T0 = wmma_bf16(aQ1, load_bt_frag_clamp(Krh, HID_, bandbase,      TK_ - 1, 32), T0);
      T1 = wmma_bf16(aQ0, load_bt_frag_clamp(Krh, HID_, bandbase + 16, TK_ - 1, 0),  T1);
      T1 = wmma_bf16(aQ1, load_bt_frag_clamp(Krh, HID_, bandbase + 16, TK_ - 1, 32), T1);

      // position->content: U[c'][n] = rel_q[clamp(bandbase+c')] . k_{js+n}
      v8f U0 = zero8(), U1 = zero8();
      U0 = wmma_bf16(load_a_frag_clamp(Qrh, HID_, bandbase,      TK_ - 1, 0),  bK0, U0);
      U0 = wmma_bf16(load_a_frag_clamp(Qrh, HID_, bandbase,      TK_ - 1, 32), bK1, U0);
      U1 = wmma_bf16(load_a_frag_clamp(Qrh, HID_, bandbase + 16, TK_ - 1, 0),  bK0, U1);
      U1 = wmma_bf16(load_a_frag_clamp(Qrh, HID_, bandbase + 16, TK_ - 1, 32), bK1, U1);

      // spill band tiles (C layout: element r = row mh+r, col nn)
#pragma unroll
      for (int r = 0; r < 8; ++r) {
        lsT[wave][mh + r][32 * s + nn]        = T0[r];
        lsT[wave][mh + r][32 * s + 16 + nn]   = T1[r];
        lsU[wave][32 * s + mh + r][nn]        = U0[r];
        lsU[wave][32 * s + 16 + mh + r][nn]   = U1[r];
      }
    }
    __syncthreads();

    // gather the diagonal band + scale
#pragma unroll
    for (int s = 0; s < 2; ++s) {
#pragma unroll
      for (int r = 0; r < 8; ++r) {
        int m = mh + r;
        int c = m - nn + 15;          // in [0,30]
        float t = lsT[wave][m][32 * s + c];
        float u = lsU[wave][32 * s + c][nn];
        S[s][r] = SCALE_F * (S[s][r] + t + u);
      }
    }

    // online softmax over this 32-wide j block
    float corr[8];
#pragma unroll
    for (int r = 0; r < 8; ++r) {
      float mx = fmaxf(S[0][r], S[1][r]);
#pragma unroll
      for (int d = 1; d < 16; d <<= 1) mx = fmaxf(mx, __shfl_xor(mx, d, 32));
      float nm = fmaxf(rmax[r], mx);
      corr[r]  = __expf(rmax[r] - nm);
      rmax[r]  = nm;
      float p0 = __expf(S[0][r] - nm);
      float p1 = __expf(S[1][r] - nm);
      S[0][r] = p0; S[1][r] = p1;
      float rs = p0 + p1;
#pragma unroll
      for (int d = 1; d < 16; d <<= 1) rs += __shfl_xor(rs, d, 32);
      rsum[r] = rsum[r] * corr[r] + rs;
    }
#pragma unroll
    for (int t = 0; t < 4; ++t)
#pragma unroll
      for (int r = 0; r < 8; ++r) oacc[t][r] *= corr[r];

    // stash P (C layout), then re-read as A-matrix layout
#pragma unroll
    for (int r = 0; r < 8; ++r) {
      lsP[wave][mh + r][nn]      = (bf16)S[0][r];
      lsP[wave][mh + r][16 + nn] = (bf16)S[1][r];
    }
    __syncthreads();

    v16bf aP;
    {
      int m = l & 15, hi = (l >> 4) * 8;
#pragma unroll
      for (int e = 0; e < 8; ++e) aP[e] = lsP[wave][m][hi + e];
#pragma unroll
      for (int e = 0; e < 8; ++e) aP[8 + e] = lsP[wave][m][16 + hi + e];
    }
#pragma unroll
    for (int t = 0; t < 4; ++t) {
      v16bf bV = load_b_rowmajor_frag(Vb + (size_t)j0 * HID_ + t * 16, HID_);
      oacc[t] = wmma_bf16(aP, bV, oacc[t]);
    }
    __syncthreads();   // protect LDS reuse next iteration
  }

  // normalize + store (bf16, pre-dense)
#pragma unroll
  for (int r = 0; r < 8; ++r) {
    float inv = 1.f / rsum[r];
    int row = i0 + mh + r;
#pragma unroll
    for (int t = 0; t < 4; ++t) {
      Oattn[(size_t)(b * SEQ_ + row) * HID_ + h * DH_ + t * 16 + nn] =
          (bf16)(oacc[t][r] * inv);
    }
  }
}

// ------------------------------------------------------------------- layernorm

__global__ __launch_bounds__(256)
void dsa_layernorm(const float* __restrict__ Y, const float* __restrict__ gamma,
                   const float* __restrict__ beta, float* __restrict__ out) {
  __shared__ float red[8];
  int row = blockIdx.x;
  const float* y = Y + (size_t)row * HID_;
  int t = threadIdx.x, l = t & 31, w = t >> 5;

  float s = 0.f;
  for (int i = t; i < HID_; i += 256) s += y[i];
#pragma unroll
  for (int d = 1; d < 32; d <<= 1) s += __shfl_xor(s, d, 32);
  if (l == 0) red[w] = s;
  __syncthreads();
  float tot = 0.f;
#pragma unroll
  for (int i = 0; i < 8; ++i) tot += red[i];
  float mu = tot / (float)HID_;
  __syncthreads();

  float v = 0.f;
  for (int i = t; i < HID_; i += 256) { float d = y[i] - mu; v += d * d; }
#pragma unroll
  for (int d = 1; d < 32; d <<= 1) v += __shfl_xor(v, d, 32);
  if (l == 0) red[w] = v;
  __syncthreads();
  float tv = 0.f;
#pragma unroll
  for (int i = 0; i < 8; ++i) tv += red[i];
  float inv = rsqrtf(tv / (float)HID_ + EPS_);

  for (int i = t; i < HID_; i += 256)
    out[(size_t)row * HID_ + i] = (y[i] - mu) * inv * gamma[i] + beta[i];
}

// ------------------------------------------------------------------- launcher

extern "C" void kernel_launch(void* const* d_in, const int* in_sizes, int n_in,
                              void* d_out, int out_size, void* d_ws, size_t ws_size,
                              hipStream_t stream) {
  (void)in_sizes; (void)n_in; (void)out_size; (void)ws_size;
  const float* hs    = (const float*)d_in[0];
  const float* pos   = (const float*)d_in[1];
  const float* Wq    = (const float*)d_in[2];
  const float* bq    = (const float*)d_in[3];
  const float* Wk    = (const float*)d_in[4];
  const float* bk    = (const float*)d_in[5];
  const float* Wv    = (const float*)d_in[6];
  const float* bv    = (const float*)d_in[7];
  const float* Wd    = (const float*)d_in[8];
  const float* bd    = (const float*)d_in[9];
  const float* gamma = (const float*)d_in[10];
  const float* beta  = (const float*)d_in[11];
  // d_in[12] rel_idx: unused, computed analytically as clip(i-j+512, 0, 1023).

  const size_t MROWS = (size_t)NB_ * SEQ_;   // 4096

  char* w = (char*)d_ws;
  auto alloc = [&](size_t bytes) -> char* {
    char* p = w;
    w += (bytes + 255) & ~(size_t)255;
    return p;
  };
  bf16* hsB  = (bf16*)alloc(MROWS * HID_ * sizeof(bf16));
  bf16* posB = (bf16*)alloc((size_t)TK_ * HID_ * sizeof(bf16));
  bf16* WqT  = (bf16*)alloc((size_t)HID_ * HID_ * sizeof(bf16));
  bf16* WkT  = (bf16*)alloc((size_t)HID_ * HID_ * sizeof(bf16));
  bf16* WvT  = (bf16*)alloc((size_t)HID_ * HID_ * sizeof(bf16));
  bf16* WdT  = (bf16*)alloc((size_t)HID_ * HID_ * sizeof(bf16));
  bf16* Qh   = (bf16*)alloc(MROWS * HID_ * sizeof(bf16));
  bf16* Kh   = (bf16*)alloc(MROWS * HID_ * sizeof(bf16));
  bf16* Vh   = (bf16*)alloc(MROWS * HID_ * sizeof(bf16));
  bf16* Qr   = (bf16*)alloc((size_t)TK_ * HID_ * sizeof(bf16));
  bf16* Kr   = (bf16*)alloc((size_t)TK_ * HID_ * sizeof(bf16));
  bf16* Ob   = (bf16*)alloc(MROWS * HID_ * sizeof(bf16));
  float* Y   = (float*)alloc(MROWS * HID_ * sizeof(float));

  // prep: bf16 conversions + weight transposes
  dsa_cvt_bf16<<<(unsigned)((MROWS * HID_ + 255) / 256), 256, 0, stream>>>(hs, hsB, (int)(MROWS * HID_));
  dsa_cvt_bf16<<<(TK_ * HID_ + 255) / 256, 256, 0, stream>>>(pos, posB, TK_ * HID_);
  dsa_transpose_bf16<<<HID_, 256, 0, stream>>>(Wq, WqT);
  dsa_transpose_bf16<<<HID_, 256, 0, stream>>>(Wk, WkT);
  dsa_transpose_bf16<<<HID_, 256, 0, stream>>>(Wv, WvT);
  dsa_transpose_bf16<<<HID_, 256, 0, stream>>>(Wd, WdT);

  // projections (WMMA)
  dim3 gHS((unsigned)(MROWS / 16), HID_ / 64);
  dim3 gPE(TK_ / 16, HID_ / 64);
  dsa_gemm_wmma<<<gHS, 128, 0, stream>>>(hsB,  WqT, bq, nullptr, nullptr, Qh);
  dsa_gemm_wmma<<<gHS, 128, 0, stream>>>(hsB,  WkT, bk, nullptr, nullptr, Kh);
  dsa_gemm_wmma<<<gHS, 128, 0, stream>>>(hsB,  WvT, bv, nullptr, nullptr, Vh);
  dsa_gemm_wmma<<<gPE, 128, 0, stream>>>(posB, WqT, bq, nullptr, nullptr, Qr);
  dsa_gemm_wmma<<<gPE, 128, 0, stream>>>(posB, WkT, bk, nullptr, nullptr, Kr);

  // fused disentangled attention (WMMA, flash-style)
  dsa_attn_wmma<<<NB_ * NH_ * (SEQ_ / 64), 128, 0, stream>>>(Qh, Kh, Vh, Qr, Kr, Ob);

  // dense + residual (WMMA), then layernorm
  dsa_gemm_wmma<<<gHS, 128, 0, stream>>>(Ob, WdT, bd, hs, Y, nullptr);
  dsa_layernorm<<<(unsigned)MROWS, 256, 0, stream>>>(Y, gamma, beta, (float*)d_out);
}
